// GraphAttentionLayer_30915174597174
// MI455X (gfx1250) — matrix operand order repaired and checked by reading, not compiled, not run on hardware
//
#include <hip/hip_runtime.h>

// ---------------------------------------------------------------------------
// GAT layer for MI455X (gfx1250), decomposed sparse-softmax formulation.
//   K1: Wh = h @ W           via v_wmma_f32_16x16x32_f16 (fp32 accum),
//       W staged global->LDS via async-to-LDS, then swizzled into B-fragment
//       layout so each lane reads its fragment as contiguous ds_load_b128.
//   K2: e_src/e_dst = Wh@a,  column-sum of Wh (block reduce + atomics)
//   K3: per-edge leakyrelu + row max (bit-pattern int atomicMax, m>=0)
//   K4: Z init = n*exp(-m);  acc init = exp(-m)*sumWh   (acc == d_out)
//   K5: per-edge scatter: w = exp(e-m)-exp(-m); Z += w; acc[src] += w*Wh[dst]
//   K6: out = elu(acc / Z)
// ---------------------------------------------------------------------------

typedef __attribute__((ext_vector_type(16))) _Float16 v16h;
typedef __attribute__((ext_vector_type(2)))  __fp16   v2fp16;
typedef __attribute__((ext_vector_type(8)))  float    v8f;
typedef __attribute__((ext_vector_type(4)))  int      v4i;

#define GAT_N      8192
#define GAT_INF    256
#define GAT_OUTF   64
#define GAT_EDGES  262144
#define GAT_ALPHA  0.2f

#if __has_builtin(__builtin_amdgcn_global_load_async_to_lds_b128)
#define GAT_HAVE_ASYNC_LDS 1
typedef __attribute__((address_space(1))) v4i gat_g_v4i;
typedef __attribute__((address_space(3))) v4i gat_l_v4i;
#endif

// ---------------- K1: projection GEMM via WMMA ------------------------------
// Grid: 64 blocks x 256 threads (8 waves). Wave w -> rows [blk*128 + w*16, +16).
// Each wave computes a 16x64 fp32 tile: 4 WMMA N-tiles, K-loop of 8 x 32.
__global__ void __launch_bounds__(256)
gat_wmma_gemm(const float* __restrict__ h, const float* __restrict__ W,
              float* __restrict__ Wh) {
  __shared__ float    Wf[GAT_INF * GAT_OUTF];        // 64 KB: raw W (f32)
  __shared__ _Float16 Wfrag[8 * 4 * 32 * 16];        // 32 KB: B fragments (f16)
  const int t = threadIdx.x;

  // ---- Stage raw W into LDS (async-to-LDS if toolchain exposes it) --------
#ifdef GAT_HAVE_ASYNC_LDS
  for (int i = t * 4; i < GAT_INF * GAT_OUTF; i += 256 * 4) {
    __builtin_amdgcn_global_load_async_to_lds_b128(
        (gat_g_v4i*)(uintptr_t)(W + i),
        (gat_l_v4i*)(uint32_t)(uintptr_t)(&Wf[i]),
        0, 0);
  }
#if __has_builtin(__builtin_amdgcn_s_wait_asynccnt)
  __builtin_amdgcn_s_wait_asynccnt(0);
#else
  asm volatile("s_wait_asynccnt 0" ::: "memory");
#endif
#else
  for (int i = t * 4; i < GAT_INF * GAT_OUTF; i += 256 * 4)
    *(float4*)&Wf[i] = *(const float4*)&W[i];
#endif
  __syncthreads();

  // ---- Swizzle W (f32, row-major KxN) into per-lane B fragments (f16) -----
  // Fragment row idx = (kTile*4 + nTile)*32 + lane; 16 contiguous halves each,
  // so the GEMM loop below loads each fragment as 2x ds_load_b128.
  for (int idx = t; idx < 8 * 4 * 32; idx += 256) {
    const int kTile = idx >> 7;          // K block of 32 (8 of them)
    const int nTile = (idx >> 5) & 3;    // N block of 16 (4 of them)
    const int ln    = idx & 31;
    const int kb    = (ln >> 4) * 8;     // 16-bit B layout: lane half h holds K 8h..8h+7, 16+8h..16+8h+7
    const int nc    = nTile * 16 + (ln & 15);
    _Float16* dst = &Wfrag[idx * 16];
    const float* src = &Wf[(kTile * 32 + kb) * GAT_OUTF + nc];
#pragma unroll
    for (int i = 0; i < 16; ++i) dst[i] = (_Float16)src[(i + (i >= 8 ? 8 : 0)) * GAT_OUTF];
  }
  __syncthreads();

  const int wave    = t >> 5;
  const int lane    = t & 31;
  const int rowBase = blockIdx.x * 128 + wave * 16;
  const int hiHalf  = lane >> 4;          // 0: lanes 0-15, 1: lanes 16-31
  const int kbA     = hiHalf * 8;         // 16-bit A layout K split
  const int ncol    = lane & 15;
  const int mrow    = rowBase + ncol;     // A row for this lane (M = lane%16)

  const v16h* WB = (const v16h*)Wfrag;
  v8f c0 = {}, c1 = {}, c2 = {}, c3 = {};

  for (int kt = 0; kt < 8; ++kt) {
    // A fragment (16x32 f16): lane holds K = kbA+0..7 and 16+kbA+0..7 of mrow.
    // float4 loads + packed rtz converts (v_cvt_pk_rtz_f16_f32).
    const float4* hp4 = (const float4*)(h + (size_t)mrow * GAT_INF + kt * 32 + kbA);
    const float4 x0 = hp4[0], x1 = hp4[1], x2 = hp4[4], x3 = hp4[5];
    union { v16h v; v2fp16 p[8]; } au;
    au.p[0] = __builtin_amdgcn_cvt_pkrtz(x0.x, x0.y);
    au.p[1] = __builtin_amdgcn_cvt_pkrtz(x0.z, x0.w);
    au.p[2] = __builtin_amdgcn_cvt_pkrtz(x1.x, x1.y);
    au.p[3] = __builtin_amdgcn_cvt_pkrtz(x1.z, x1.w);
    au.p[4] = __builtin_amdgcn_cvt_pkrtz(x2.x, x2.y);
    au.p[5] = __builtin_amdgcn_cvt_pkrtz(x2.z, x2.w);
    au.p[6] = __builtin_amdgcn_cvt_pkrtz(x3.x, x3.y);
    au.p[7] = __builtin_amdgcn_cvt_pkrtz(x3.z, x3.w);

    // B fragments: contiguous 32-byte chunks per lane.
    const v16h* wb = WB + kt * 128 + lane;
    const v16h b0 = wb[0];
    const v16h b1 = wb[32];
    const v16h b2 = wb[64];
    const v16h b3 = wb[96];

    c0 = __builtin_amdgcn_wmma_f32_16x16x32_f16(false, au.v, false, b0, (short)0, c0, false, false);
    c1 = __builtin_amdgcn_wmma_f32_16x16x32_f16(false, au.v, false, b1, (short)0, c1, false, false);
    c2 = __builtin_amdgcn_wmma_f32_16x16x32_f16(false, au.v, false, b2, (short)0, c2, false, false);
    c3 = __builtin_amdgcn_wmma_f32_16x16x32_f16(false, au.v, false, b3, (short)0, c3, false, false);
  }

  // D layout: VGPR r holds row (r + 8*hiHalf), column = ncol
#pragma unroll
  for (int r = 0; r < 8; ++r) {
    const int row = rowBase + r + hiHalf * 8;
    float* op = Wh + (size_t)row * GAT_OUTF + ncol;
    op[0] = c0[r];  op[16] = c1[r];  op[32] = c2[r];  op[48] = c3[r];
  }
}

// ---------------- K2: node scores + column sum of Wh ------------------------
__global__ void __launch_bounds__(256)
gat_node_scores(const float* __restrict__ Wh, const float* __restrict__ a,
                float* __restrict__ e_src, float* __restrict__ e_dst,
                float* __restrict__ sumWh) {
  __shared__ float sred[4][GAT_OUTF];
  const int t   = threadIdx.x;
  const int row = blockIdx.x * 256 + t;

  const float* wr = Wh + (size_t)row * GAT_OUTF;
  float ds = 0.f, dd = 0.f;
#pragma unroll 8
  for (int c = 0; c < GAT_OUTF; ++c) {
    const float v = wr[c];
    ds += v * a[c];
    dd += v * a[GAT_OUTF + c];
  }
  e_src[row] = ds;
  e_dst[row] = dd;

  // column partial sums: 4 groups of 64 threads, each strides 256 rows
  const int c = t & 63, g = t >> 6;
  const int base = blockIdx.x * 256;
  float p = 0.f;
  for (int r = g; r < 256; r += 4)
    p += Wh[(size_t)(base + r) * GAT_OUTF + c];
  sred[g][c] = p;
  __syncthreads();
  if (t < GAT_OUTF)
    atomicAdd(&sumWh[t], sred[0][t] + sred[1][t] + sred[2][t] + sred[3][t]);
}

// ---------------- K3: edge leakyrelu + row max ------------------------------
__global__ void __launch_bounds__(256)
gat_edge_max(const int* __restrict__ ei, const float* __restrict__ e_src,
             const float* __restrict__ e_dst, float* __restrict__ e_edge,
             float* __restrict__ m) {
  const int e = blockIdx.x * 256 + threadIdx.x;
  if (e >= GAT_EDGES) return;
  const int s = ei[e], d = ei[GAT_EDGES + e];
  float v = e_src[s] + e_dst[d];
  v = v > 0.f ? v : GAT_ALPHA * v;
  e_edge[e] = v;
  // m initialized to 0.0f; for candidates > 0 int-bits order == float order,
  // negative candidates map to negative ints and correctly never win.
  atomicMax((int*)&m[s], __float_as_int(v));
}

// ---------------- K4: per-node init of Z and acc (== d_out) -----------------
__global__ void __launch_bounds__(256)
gat_node_init(const float* __restrict__ m, const float* __restrict__ sumWh,
              float* __restrict__ Z, float* __restrict__ acc) {
  const int i = blockIdx.x * 256 + threadIdx.x;
  if (i >= GAT_N) return;
  const float em = __expf(-m[i]);
  Z[i] = (float)GAT_N * em;                 // n * exp(-m): all-zeros baseline
  float* ap = acc + (size_t)i * GAT_OUTF;
#pragma unroll 8
  for (int c = 0; c < GAT_OUTF; ++c)
    ap[c] = em * sumWh[c];                  // exp(-m) * sum_j Wh[j]
}

// ---------------- K5: per-edge scatter (16 lanes / edge) --------------------
__global__ void __launch_bounds__(256)
gat_edge_scatter(const int* __restrict__ ei, const float* __restrict__ e_edge,
                 const float* __restrict__ m, const float* __restrict__ Wh,
                 float* __restrict__ Z, float* __restrict__ acc) {
  const int t = blockIdx.x * 256 + threadIdx.x;
  const int e = t >> 4;
  const int q = t & 15;
  if (e >= GAT_EDGES) return;
  const int s = ei[e], d = ei[GAT_EDGES + e];
  const float ms = m[s];
  const float w  = __expf(e_edge[e] - ms) - __expf(-ms);
  if (q == 0) atomicAdd(&Z[s], w);
  const float4 wh4 = ((const float4*)(Wh + (size_t)d * GAT_OUTF))[q];
  float* ap = acc + (size_t)s * GAT_OUTF + q * 4;
  atomicAdd(ap + 0, w * wh4.x);
  atomicAdd(ap + 1, w * wh4.y);
  atomicAdd(ap + 2, w * wh4.z);
  atomicAdd(ap + 3, w * wh4.w);
}

// ---------------- K6: normalize + ELU ---------------------------------------
__global__ void __launch_bounds__(256)
gat_finalize(float* __restrict__ acc, const float* __restrict__ Z) {
  const int t = blockIdx.x * 256 + threadIdx.x;
  if (t >= GAT_N * GAT_OUTF) return;
  const float v = acc[t] / Z[t >> 6];
  acc[t] = v > 0.f ? v : (__expf(v) - 1.f);
}

// ---------------------------------------------------------------------------
extern "C" void kernel_launch(void* const* d_in, const int* in_sizes, int n_in,
                              void* d_out, int out_size, void* d_ws, size_t ws_size,
                              hipStream_t stream) {
  const float* h  = (const float*)d_in[0];   // [8192, 256]
  const int*   ei = (const int*)d_in[1];     // [2, 262144]
  const float* W  = (const float*)d_in[2];   // [256, 64]
  const float* a  = (const float*)d_in[3];   // [128, 1]
  float* out = (float*)d_out;                // [8192, 64], doubles as acc

  float* ws     = (float*)d_ws;
  float* Wh     = ws;                              // 524288 floats
  float* e_src  = Wh + (size_t)GAT_N * GAT_OUTF;   // 8192
  float* e_dst  = e_src + GAT_N;                   // 8192
  float* m      = e_dst + GAT_N;                   // 8192
  float* Z      = m + GAT_N;                       // 8192
  float* sumWh  = Z + GAT_N;                       // 64
  float* e_edge = sumWh + GAT_OUTF;                // 262144
  // total ~3.1 MB of workspace

  (void)hipMemsetAsync(m,     0, GAT_N * sizeof(float), stream);
  (void)hipMemsetAsync(sumWh, 0, GAT_OUTF * sizeof(float), stream);

  gat_wmma_gemm   <<<GAT_N / 128, 256, 0, stream>>>(h, W, Wh);
  gat_node_scores <<<GAT_N / 256, 256, 0, stream>>>(Wh, a, e_src, e_dst, sumWh);
  gat_edge_max    <<<GAT_EDGES / 256, 256, 0, stream>>>(ei, e_src, e_dst, e_edge, m);
  gat_node_init   <<<GAT_N / 256, 256, 0, stream>>>(m, sumWh, Z, out);
  gat_edge_scatter<<<GAT_EDGES * 16 / 256, 256, 0, stream>>>(ei, e_edge, m, Wh, Z, out);
  gat_finalize    <<<GAT_N * GAT_OUTF / 256, 256, 0, stream>>>(out, Z);
}
